// VectorQuantizer_3530463117332
// MI455X (gfx1250) — matrix operand order repaired and checked by reading, not compiled, not run on hardware
//
#include <hip/hip_runtime.h>
#include <hip/hip_bf16.h>
#include <math.h>

// ---------------------------------------------------------------------------
// VQ-VAE vector quantizer for MI455X (gfx1250, wave32, WMMA).
//   N = 32768 rows, D = 256, K = 8192 codes.
//   argmin_k ( ||e_k||^2 - 2 * x.e_k )   via bf16 WMMA GEMM, fused argmin.
//   quantized_st == gather(codebook, idx); loss = 1.25 * mean((q-x)^2).
//
// Round 2: 32-column tiles, double-buffered LDS (1 barrier/tile), dual
// accumulator chains so WMMA->VALU hazard slots are filled by the second
// chain. 16 v_wmma per barrier phase.
// ---------------------------------------------------------------------------

#define VQ_N 32768
#define VQ_D 256
#define VQ_K 8192
#define TILE_PAD 264            // 256 + 8 bf16 padding per LDS row
#define TILE_COLS 32            // codes per tile
#define N_TILES (VQ_K / TILE_COLS)   // 256

typedef __attribute__((ext_vector_type(16))) __bf16 v16bf;
typedef __attribute__((ext_vector_type(8)))  __bf16 v8bf;
typedef __attribute__((ext_vector_type(8)))  float  v8f;

// ---------------------------------------------------------------------------
// Kernel 1: codebook f32 -> bf16, plus per-code squared norms.
// ---------------------------------------------------------------------------
__global__ __launch_bounds__(256) void vq_prep_codebook(
    const float* __restrict__ cb, __bf16* __restrict__ cbb,
    float* __restrict__ cbnorm) {
  const int k = blockIdx.x;
  const int d = threadIdx.x;
  const float v = cb[(size_t)k * VQ_D + d];
  cbb[(size_t)k * VQ_D + d] = (__bf16)v;
  __shared__ float red[256];
  red[d] = v * v;
  __syncthreads();
  for (int s = 128; s > 0; s >>= 1) {
    if (d < s) red[d] += red[d + s];
    __syncthreads();
  }
  if (d == 0) cbnorm[k] = red[0];
}

// ---------------------------------------------------------------------------
// Kernel 2: zero counts + sse accumulator.
// ---------------------------------------------------------------------------
__global__ void vq_init(int* __restrict__ counts, float* __restrict__ sse) {
  const int i = blockIdx.x * blockDim.x + threadIdx.x;
  if (i < VQ_K) counts[i] = 0;
  if (i == 0) *sse = 0.0f;
}

// ---------------------------------------------------------------------------
// Kernel 3: fused bf16-WMMA distance GEMM + running argmin.
//   Grid: 256 WGs x 256 threads (8 waves). Wave w owns rows
//   blockIdx.x*128 + w*16 .. +15; A fragments resident in registers.
//   32-code B tiles double-buffered in LDS; two accumulator chains.
// ---------------------------------------------------------------------------
__global__ __launch_bounds__(256) void vq_argmin_kernel(
    const float* __restrict__ x, const __bf16* __restrict__ cb,
    const float* __restrict__ cbnorm, int* __restrict__ idx_out) {
  __shared__ __attribute__((aligned(16))) __bf16 smem[2][TILE_COLS * TILE_PAD];

  const int t    = threadIdx.x;
  const int wave = t >> 5;
  const int lane = t & 31;
  const int hf   = lane >> 4;   // half-wave: 0 or 1
  const int n    = lane & 15;   // column-in-subtile / A row-in-wave
  const int row_base = blockIdx.x * 128 + wave * 16;

  // ---- Resident A fragments: 16 rows x 256 (f32 -> bf16 on the fly).
  // ISA 16-bit A layout: elem e<8  -> K = kk*32 + 8*hf + e
  //                      elem e>=8 -> K = kk*32 + 16 + 8*hf + (e-8)
  v16bf a[8];
  {
    const float* xr = x + (size_t)(row_base + n) * VQ_D;
#pragma unroll
    for (int kk = 0; kk < 8; ++kk) {
      const int d0 = kk * 32 + hf * 8;
      const float4 f0 = *(const float4*)(xr + d0);
      const float4 f1 = *(const float4*)(xr + d0 + 4);
      const float4 f2 = *(const float4*)(xr + d0 + 16);
      const float4 f3 = *(const float4*)(xr + d0 + 20);
      v16bf af;
      af[0]  = (__bf16)f0.x; af[1]  = (__bf16)f0.y;
      af[2]  = (__bf16)f0.z; af[3]  = (__bf16)f0.w;
      af[4]  = (__bf16)f1.x; af[5]  = (__bf16)f1.y;
      af[6]  = (__bf16)f1.z; af[7]  = (__bf16)f1.w;
      af[8]  = (__bf16)f2.x; af[9]  = (__bf16)f2.y;
      af[10] = (__bf16)f2.z; af[11] = (__bf16)f2.w;
      af[12] = (__bf16)f3.x; af[13] = (__bf16)f3.y;
      af[14] = (__bf16)f3.z; af[15] = (__bf16)f3.w;
      a[kk] = af;
    }
  }

  // Running per-lane argmin state: acc element v <-> row m = 8*hf + v.
  float bestv[8];
  int   besti[8];
#pragma unroll
  for (int v = 0; v < 8; ++v) { bestv[v] = 3.4e38f; besti[v] = 0; }

  // Cooperative tile load: 32 rows x 256 bf16 = 16 KB; each of the 256
  // threads moves 64 B (4 x b128) per tile.
  const int ldr = t >> 3;          // tile row 0..31
  const int ldc = (t & 7) * 32;    // element offset 0..224
  uint4 pre0, pre1, pre2, pre3;

  {
    const __bf16* src = cb + (size_t)ldr * VQ_D + ldc;   // tile 0
    pre0 = *(const uint4*)(src);
    pre1 = *(const uint4*)(src + 8);
    pre2 = *(const uint4*)(src + 16);
    pre3 = *(const uint4*)(src + 24);
    __bf16* dst = &smem[0][ldr * TILE_PAD + ldc];
    *(uint4*)(dst)      = pre0;
    *(uint4*)(dst + 8)  = pre1;
    *(uint4*)(dst + 16) = pre2;
    *(uint4*)(dst + 24) = pre3;
  }

  for (int ct = 0; ct < N_TILES; ++ct) {
    __syncthreads();   // buf[ct&1] written; buf[(ct+1)&1] free for reuse
    const int buf = ct & 1;

    if (ct + 1 < N_TILES) {   // register-prefetch next tile
      const __bf16* src = cb + (size_t)(ct + 1) * TILE_COLS * VQ_D
                          + (size_t)ldr * VQ_D + ldc;
      pre0 = *(const uint4*)(src);
      pre1 = *(const uint4*)(src + 8);
      pre2 = *(const uint4*)(src + 16);
      pre3 = *(const uint4*)(src + 24);
    }

    const float norm0 = cbnorm[ct * TILE_COLS + n];
    const float norm1 = cbnorm[ct * TILE_COLS + 16 + n];

    v8f acc0 = {0.f, 0.f, 0.f, 0.f, 0.f, 0.f, 0.f, 0.f};
    v8f acc1 = {0.f, 0.f, 0.f, 0.f, 0.f, 0.f, 0.f, 0.f};
#pragma unroll
    for (int kk = 0; kk < 8; ++kk) {
      // ISA 16-bit B layout (32x16): elem e<8 -> K = kk*32 + 8*hf + e, e>=8 -> K = kk*32 + 16 + 8*hf + (e-8)
      const int dbase = kk * 32 + 8 * hf;
      const v8bf lo0 = *(const v8bf*)(&smem[buf][n * TILE_PAD + dbase]);
      const v8bf hi0 = *(const v8bf*)(&smem[buf][n * TILE_PAD + dbase + 16]);
      const v8bf lo1 = *(const v8bf*)(&smem[buf][(n + 16) * TILE_PAD + dbase]);
      const v8bf hi1 = *(const v8bf*)(&smem[buf][(n + 16) * TILE_PAD + dbase + 16]);
      const v16bf b0 = __builtin_shufflevector(lo0, hi0,
          0, 1, 2, 3, 4, 5, 6, 7, 8, 9, 10, 11, 12, 13, 14, 15);
      const v16bf b1 = __builtin_shufflevector(lo1, hi1,
          0, 1, 2, 3, 4, 5, 6, 7, 8, 9, 10, 11, 12, 13, 14, 15);
      acc0 = __builtin_amdgcn_wmma_f32_16x16x32_bf16(
          false, a[kk], false, b0, (short)0, acc0, false, false);
      acc1 = __builtin_amdgcn_wmma_f32_16x16x32_bf16(
          false, a[kk], false, b1, (short)0, acc1, false, false);
    }

    const int k0 = ct * TILE_COLS + n;
    const int k1 = k0 + 16;
#pragma unroll
    for (int v = 0; v < 8; ++v) {
      const float d0 = fmaf(-2.0f, acc0[v], norm0);
      if (d0 < bestv[v]) { bestv[v] = d0; besti[v] = k0; }
      const float d1 = fmaf(-2.0f, acc1[v], norm1);
      if (d1 < bestv[v]) { bestv[v] = d1; besti[v] = k1; }
    }

    if (ct + 1 < N_TILES) {   // stage next tile into the other buffer
      __bf16* dst = &smem[buf ^ 1][ldr * TILE_PAD + ldc];
      *(uint4*)(dst)      = pre0;
      *(uint4*)(dst + 8)  = pre1;
      *(uint4*)(dst + 16) = pre2;
      *(uint4*)(dst + 24) = pre3;
    }
  }

  // Reduce (min, argmin) across the 16 lanes of each half-wave.
#pragma unroll
  for (int off = 1; off < 16; off <<= 1) {
#pragma unroll
    for (int v = 0; v < 8; ++v) {
      const float ov = __shfl_xor(bestv[v], off, 32);
      const int   oi = __shfl_xor(besti[v], off, 32);
      if (ov < bestv[v] || (ov == bestv[v] && oi < besti[v])) {
        bestv[v] = ov; besti[v] = oi;
      }
    }
  }

  if (n == 0) {
#pragma unroll
    for (int v = 0; v < 8; ++v)
      idx_out[row_base + 8 * hf + v] = besti[v];
  }
}

// ---------------------------------------------------------------------------
// Kernel 4: gather quantized rows, SSE reduction, index histogram.
// ---------------------------------------------------------------------------
__global__ __launch_bounds__(256) void vq_gather_loss(
    const float* __restrict__ x, const float* __restrict__ cb,
    const int* __restrict__ idx, float* __restrict__ out,
    float* __restrict__ sse, int* __restrict__ counts) {
  const int nrow = blockIdx.x;
  const int d = threadIdx.x;
  const int k = idx[nrow];
  const float q  = cb[(size_t)k * VQ_D + d];
  const float xv = x[(size_t)nrow * VQ_D + d];
  out[(size_t)nrow * VQ_D + d] = q;       // quantized_st value == quantized
  const float diff = q - xv;
  __shared__ float red[256];
  red[d] = diff * diff;
  __syncthreads();
  for (int s = 128; s > 0; s >>= 1) {
    if (d < s) red[d] += red[d + s];
    __syncthreads();
  }
  if (d == 0) {
    atomicAdd(sse, red[0]);
    atomicAdd(&counts[k], 1);
  }
}

// ---------------------------------------------------------------------------
// Kernel 5: loss + perplexity scalars.
// ---------------------------------------------------------------------------
__global__ __launch_bounds__(256) void vq_finalize(
    const int* __restrict__ counts, const float* __restrict__ sse,
    float* __restrict__ out) {
  __shared__ float red[256];
  const int t = threadIdx.x;
  float local = 0.0f;
  for (int k = t; k < VQ_K; k += 256) {
    const float p = (float)counts[k] * (1.0f / (float)VQ_N);
    local += p * logf(p + 1e-10f);
  }
  red[t] = local;
  __syncthreads();
  for (int s = 128; s > 0; s >>= 1) {
    if (t < s) red[t] += red[t + s];
    __syncthreads();
  }
  if (t == 0) {
    out[(size_t)VQ_N * VQ_D]     = 1.25f * (*sse) * (1.0f / ((float)VQ_N * (float)VQ_D));
    out[(size_t)VQ_N * VQ_D + 1] = expf(-red[0]);
  }
}

// ---------------------------------------------------------------------------
extern "C" void kernel_launch(void* const* d_in, const int* in_sizes, int n_in,
                              void* d_out, int out_size, void* d_ws, size_t ws_size,
                              hipStream_t stream) {
  const float* x  = (const float*)d_in[0];   // [32768, 256] f32
  const float* cb = (const float*)d_in[1];   // [8192, 256] f32
  float* out = (float*)d_out;                // [32768*256 + 2] f32

  // Workspace layout (bytes):
  char* ws = (char*)d_ws;
  __bf16* cbb  = (__bf16*)(ws);                                      // 4 MB
  float*  cbn  = (float*)(ws + (size_t)VQ_K * VQ_D * 2);             // 32 KB
  int*    idx  = (int*)(ws + (size_t)VQ_K * VQ_D * 2 + VQ_K * 4);    // 128 KB
  int*    cnts = (int*)((char*)idx + (size_t)VQ_N * 4);              // 32 KB
  float*  sse  = (float*)((char*)cnts + (size_t)VQ_K * 4);           // 4 B

  vq_prep_codebook<<<VQ_K, 256, 0, stream>>>(cb, cbb, cbn);
  vq_init<<<(VQ_K + 255) / 256, 256, 0, stream>>>(cnts, sse);
  vq_argmin_kernel<<<VQ_N / 128, 256, 0, stream>>>(x, cbb, cbn, idx);
  vq_gather_loss<<<VQ_N, 256, 0, stream>>>(x, cb, idx, out, sse, cnts);
  vq_finalize<<<1, 256, 0, stream>>>(cnts, sse, out);
}
